// MLPBlock_24833500906097
// MI455X (gfx1250) — matrix operand order repaired and checked by reading, not compiled, not run on hardware
//
#include <hip/hip_runtime.h>
#include <hip/hip_bf16.h>
#include <math.h>

// ---- problem constants ----
#define KT 4096      // tokens
#define KD 2048      // hidden
#define KF 4096      // ffn
#define KE 8         // experts
#define KK 2         // top-k
#define KALPHA 1.702f
#define KLIMIT 7.0f

typedef __bf16 bf16_t;
typedef __attribute__((ext_vector_type(16))) __bf16      v16bf;
typedef __attribute__((ext_vector_type(8)))  float       v8f;
typedef __attribute__((ext_vector_type(4)))  unsigned int u32x4;

union FragB16 { v16bf v; u32x4 q[2]; };

// LDS tile rows: 64 bf16 of data (128B) padded to 144B to spread banks.
#define ROWB 144

// A fragment (16-bit A 16x32, ISA 7.12.2): lane m=lane&15, hi=lane>>4.
//   VGPR0-3: K = hi*8 + 0..7  -> bytes row*ROWB + kc2 + hi*16
//   VGPR4-7: K = 16 + hi*8 .. -> +32
__device__ __forceinline__ v16bf lds_frag_a(const bf16_t* tile, int row, int kc2, int hi) {
    const char* p = (const char*)tile + row * ROWB + kc2 + hi * 16;
    FragB16 f;
    f.q[0] = *(const u32x4*)p;
    f.q[1] = *(const u32x4*)(p + 32);
    return f.v;
}
// B fragment (16-bit B 32x16): lane n=lane&15, hi=lane>>4.
//   VGPR0-3: K = hi*8 + 0..7 -> bytes row*ROWB + kc2 + hi*16; VGPR4-7: K = 16 + hi*8 .. -> +32
__device__ __forceinline__ v16bf lds_frag_b(const bf16_t* tile, int row, int kc2, int hi) {
    const char* p = (const char*)tile + row * ROWB + kc2 + hi * 16;
    FragB16 f;
    f.q[0] = *(const u32x4*)p;
    f.q[1] = *(const u32x4*)(p + 32);
    return f.v;
}

// gfx1250 async global->LDS copy (16B per lane), tracked by ASYNCcnt.
__device__ __forceinline__ void async_ld16(unsigned lds_addr, const void* gaddr) {
    asm volatile("global_load_async_to_lds_b128 %0, %1, off"
                 :: "v"(lds_addr), "v"(gaddr) : "memory");
}
template <int N>
__device__ __forceinline__ void wait_asynccnt() {
    asm volatile("s_wait_asynccnt %0" :: "i"(N) : "memory");
}
__device__ __forceinline__ unsigned lds_off(const void* p) {
    return (unsigned)(size_t)p;   // generic LDS pointer: low 32 bits = LDS byte address
}

// ---------------- x -> bf16 ----------------
__global__ __launch_bounds__(256) void cvtx_kernel(const float* __restrict__ x,
                                                   bf16_t* __restrict__ xb) {
    size_t idx = (size_t)blockIdx.x * 256 + threadIdx.x;   // each converts 8 floats
    const float4* s = (const float4*)x;
    float4 f0 = s[idx * 2];
    float4 f1 = s[idx * 2 + 1];
    union { bf16_t h[8]; u32x4 q; } u;
    u.h[0] = (bf16_t)f0.x; u.h[1] = (bf16_t)f0.y; u.h[2] = (bf16_t)f0.z; u.h[3] = (bf16_t)f0.w;
    u.h[4] = (bf16_t)f1.x; u.h[5] = (bf16_t)f1.y; u.h[6] = (bf16_t)f1.z; u.h[7] = (bf16_t)f1.w;
    *((u32x4*)xb + idx) = u.q;
}

// ------- per-expert transpose + fp32->bf16: src [R][C] -> dst [C][R] -------
__global__ __launch_bounds__(256) void transcvt_kernel(const float* __restrict__ src,
                                                       bf16_t* __restrict__ dst,
                                                       int R, int C) {
    __shared__ float tile[32][33];
    int e = blockIdx.z;
    src += (size_t)e * R * C;
    dst += (size_t)e * R * C;
    int c0 = blockIdx.x * 32, r0 = blockIdx.y * 32;
    int tx = threadIdx.x & 31, ty = threadIdx.x >> 5;
#pragma unroll
    for (int i = 0; i < 4; i++)
        tile[ty + i * 8][tx] = src[(size_t)(r0 + ty + i * 8) * C + c0 + tx];
    __syncthreads();
#pragma unroll
    for (int i = 0; i < 4; i++)
        dst[(size_t)(c0 + ty + i * 8) * R + r0 + tx] = (bf16_t)tile[tx][ty + i * 8];
}

// ---------------- router: one wave32 per token ----------------
__global__ __launch_bounds__(256) void router_kernel(const float* __restrict__ x,
                                                     const float* __restrict__ rw,
                                                     const float* __restrict__ rb,
                                                     int* __restrict__ topk_idx,
                                                     float* __restrict__ topk_w) {
    int wid = threadIdx.x >> 5, lane = threadIdx.x & 31;
    int t = blockIdx.x * 8 + wid;
    const float4* xr = (const float4*)(x + (size_t)t * KD);
    float acc[KE];
#pragma unroll
    for (int e = 0; e < KE; e++) acc[e] = 0.0f;
    for (int c = lane; c < KD / 4; c += 32) {
        float4 xv = xr[c];
#pragma unroll
        for (int e = 0; e < KE; e++) {
            float4 wv = ((const float4*)(rw + (size_t)e * KD))[c];
            acc[e] += xv.x * wv.x + xv.y * wv.y + xv.z * wv.z + xv.w * wv.w;
        }
    }
#pragma unroll
    for (int e = 0; e < KE; e++)
        for (int m = 16; m > 0; m >>= 1) acc[e] += __shfl_xor(acc[e], m, 32);
    if (lane == 0) {
        float lg[KE], mx = -1e30f;
#pragma unroll
        for (int e = 0; e < KE; e++) { lg[e] = acc[e] + rb[e]; mx = fmaxf(mx, lg[e]); }
        float p[KE], s = 0.0f;
#pragma unroll
        for (int e = 0; e < KE; e++) { p[e] = __expf(lg[e] - mx); s += p[e]; }
        float inv = __builtin_amdgcn_rcpf(s);
#pragma unroll
        for (int e = 0; e < KE; e++) p[e] *= inv;
        int e0 = 0; float p0 = p[0];
#pragma unroll
        for (int e = 1; e < KE; e++) if (p[e] > p0) { p0 = p[e]; e0 = e; }
        int e1 = -1; float p1 = -1e30f;
#pragma unroll
        for (int e = 0; e < KE; e++) if (e != e0 && p[e] > p1) { p1 = p[e]; e1 = e; }
        float rsum = 1.0f / (p0 + p1);
        topk_idx[2 * t]     = e0;  topk_w[2 * t]     = p0 * rsum;
        topk_idx[2 * t + 1] = e1;  topk_w[2 * t + 1] = p1 * rsum;
    }
}

// -------- deterministic stable counting sort of (token,k) slots by expert --------
__global__ __launch_bounds__(256) void sortk_kernel(const int* __restrict__ topk_idx,
                                                    int* __restrict__ perm,
                                                    int* __restrict__ slot_of,
                                                    int* __restrict__ counts,
                                                    int* __restrict__ offsets) {
    __shared__ int cnt[256][KE];
    __shared__ int tot[KE];
    __shared__ int base[KE];
    int tid = threadIdx.x;
    int myidx[32];
    int lc[KE];
#pragma unroll
    for (int e = 0; e < KE; e++) lc[e] = 0;
#pragma unroll
    for (int q = 0; q < 32; q++) {
        int i = tid * 32 + q;
        myidx[q] = topk_idx[i];
        lc[myidx[q]]++;
    }
#pragma unroll
    for (int e = 0; e < KE; e++) cnt[tid][e] = lc[e];
    __syncthreads();
    if (tid < KE) {
        int s = 0;
        for (int j = 0; j < 256; j++) s += cnt[j][tid];
        tot[tid] = s;
    }
    __syncthreads();
    if (tid == 0) {
        int run = 0;
        for (int e = 0; e < KE; e++) {
            base[e] = run;
            counts[e] = tot[e];
            offsets[e] = run;
            run += tot[e];
        }
    }
    __syncthreads();
    int start[KE];
#pragma unroll
    for (int e = 0; e < KE; e++) {
        int s = base[e];
        for (int j = 0; j < tid; j++) s += cnt[j][e];
        start[e] = s;
    }
#pragma unroll
    for (int q = 0; q < 32; q++) {
        int i = tid * 32 + q;
        int e = myidx[q];
        int pos = start[e]++;
        perm[pos] = i >> 1;      // token id
        slot_of[i] = pos;
    }
}

// -------- GEMM1: h = swiglu(x@w1+b1, x@w3+b3) per expert, gathered rows --------
// block tile: 128 (slots) x 64 (F). K staged 64 at a time, double-buffered via
// async global->LDS copies (ASYNCcnt); 8 waves = 4x2 wave grid, 16 WMMA/stage.
__global__ __launch_bounds__(256) void gemm1_kernel(const bf16_t* __restrict__ xb,
                                                    const bf16_t* __restrict__ w1t,
                                                    const bf16_t* __restrict__ w3t,
                                                    const float* __restrict__ b1,
                                                    const float* __restrict__ b3,
                                                    const int* __restrict__ counts,
                                                    const int* __restrict__ offsets,
                                                    const int* __restrict__ perm,
                                                    bf16_t* __restrict__ h) {
    int e = blockIdx.z;
    int cnt = counts[e];
    int m0 = blockIdx.y * 128;
    if (m0 >= cnt) return;
    int off = offsets[e];
    int n0 = blockIdx.x * 64;
    int tid = threadIdx.x;
    int lane = tid & 31, wid = tid >> 5;
    int wr = wid & 3, wc = wid >> 2;
    int hi = lane >> 4, ln = lane & 15;

    __shared__ __align__(16) bf16_t As[2][128 * (ROWB / 2)];
    __shared__ __align__(16) bf16_t B1s[2][64 * (ROWB / 2)];
    __shared__ __align__(16) bf16_t B3s[2][64 * (ROWB / 2)];

    // staging assignments (per stage: 64 K-values per row)
    int ar = tid >> 1, ah = tid & 1;                 // A: row, 64B half -> 4x16B
    int arow_g = m0 + ar;
    int tokr = (arow_g < cnt) ? perm[off + arow_g] : 0;
    const bf16_t* aSrc = xb + (size_t)tokr * KD + ah * 32;
    unsigned aDst0 = lds_off(&As[0][0]) + ar * ROWB + ah * 64;
    unsigned aDst1 = lds_off(&As[1][0]) + ar * ROWB + ah * 64;
    int br = tid >> 2, bq = tid & 3;                 // B: row, 32B quarter -> 2x16B
    const bf16_t* b1Src = w1t + ((size_t)e * KF + (n0 + br)) * KD + bq * 16;
    const bf16_t* b3Src = w3t + ((size_t)e * KF + (n0 + br)) * KD + bq * 16;
    unsigned b1Dst0 = lds_off(&B1s[0][0]) + br * ROWB + bq * 32;
    unsigned b1Dst1 = lds_off(&B1s[1][0]) + br * ROWB + bq * 32;
    unsigned b3Dst0 = lds_off(&B3s[0][0]) + br * ROWB + bq * 32;
    unsigned b3Dst1 = lds_off(&B3s[1][0]) + br * ROWB + bq * 32;

    auto issue_stage = [&](int k0, int b) {
        unsigned ad = b ? aDst1 : aDst0;
        unsigned b1d = b ? b1Dst1 : b1Dst0;
        unsigned b3d = b ? b3Dst1 : b3Dst0;
#pragma unroll
        for (int q = 0; q < 4; q++) async_ld16(ad + q * 16, aSrc + k0 + q * 8);
#pragma unroll
        for (int q = 0; q < 2; q++) async_ld16(b1d + q * 16, b1Src + k0 + q * 8);
#pragma unroll
        for (int q = 0; q < 2; q++) async_ld16(b3d + q * 16, b3Src + k0 + q * 8);
    };

    v8f cg[2][2], cu[2][2];
#pragma unroll
    for (int i = 0; i < 2; i++)
#pragma unroll
        for (int j = 0; j < 2; j++) { cg[i][j] = (v8f)(0.0f); cu[i][j] = (v8f)(0.0f); }

    const int NST = KD / 64;
    issue_stage(0, 0);
    for (int s = 0; s < NST; s++) {
        __syncthreads();                       // all waves done reading buf[(s+1)&1]
        if (s + 1 < NST) {
            issue_stage((s + 1) * 64, (s + 1) & 1);
            wait_asynccnt<8>();                // stage s landed (8 newest still fly)
        } else {
            wait_asynccnt<0>();
        }
        __syncthreads();                       // stage s visible to all waves
        const bf16_t* Ab = As[s & 1];
        const bf16_t* B1b = B1s[s & 1];
        const bf16_t* B3b = B3s[s & 1];
#pragma unroll
        for (int kc = 0; kc < 2; kc++) {
            int kc2 = kc * 64;                 // byte offset of 32-wide k-chunk
            v16bf a[2], bg[2], bu[2];
#pragma unroll
            for (int i = 0; i < 2; i++) a[i] = lds_frag_a(Ab, wr * 32 + i * 16 + ln, kc2, hi);
#pragma unroll
            for (int j = 0; j < 2; j++) {
                bg[j] = lds_frag_b(B1b, wc * 32 + j * 16 + ln, kc2, hi);
                bu[j] = lds_frag_b(B3b, wc * 32 + j * 16 + ln, kc2, hi);
            }
#pragma unroll
            for (int i = 0; i < 2; i++)
#pragma unroll
                for (int j = 0; j < 2; j++) {
                    cg[i][j] = __builtin_amdgcn_wmma_f32_16x16x32_bf16(
                        false, a[i], false, bg[j], (short)0, cg[i][j], false, false);
                    cu[i][j] = __builtin_amdgcn_wmma_f32_16x16x32_bf16(
                        false, a[i], false, bu[j], (short)0, cu[i][j], false, false);
                }
        }
    }

    // epilogue: bias + clamp + swiglu, store bf16 h
#pragma unroll
    for (int j = 0; j < 2; j++) {
        int nf = n0 + wc * 32 + j * 16 + ln;
        float b1v = b1[e * KF + nf];
        float b3v = b3[e * KF + nf];
#pragma unroll
        for (int i = 0; i < 2; i++) {
#pragma unroll
            for (int k = 0; k < 8; k++) {
                int mg = m0 + wr * 32 + i * 16 + k + hi * 8;
                if (mg < cnt) {
                    float gv = cg[i][j][k] + b1v;
                    float uv = cu[i][j][k] + b3v;
                    gv = fminf(gv, KLIMIT);
                    uv = fminf(fmaxf(uv, -KLIMIT), KLIMIT);
                    float glu = gv * __builtin_amdgcn_rcpf(1.0f + __expf(-KALPHA * gv));
                    float hv = (uv + 1.0f) * glu;
                    h[(size_t)(off + mg) * KF + nf] = (bf16_t)hv;
                }
            }
        }
    }
}

// -------- GEMM2: y = h @ w2 per expert (bias applied in combine) --------
__global__ __launch_bounds__(256) void gemm2_kernel(const bf16_t* __restrict__ h,
                                                    const bf16_t* __restrict__ w2t,
                                                    const int* __restrict__ counts,
                                                    const int* __restrict__ offsets,
                                                    float* __restrict__ y) {
    int e = blockIdx.z;
    int cnt = counts[e];
    int m0 = blockIdx.y * 128;
    if (m0 >= cnt) return;
    int off = offsets[e];
    int n0 = blockIdx.x * 64;
    int tid = threadIdx.x;
    int lane = tid & 31, wid = tid >> 5;
    int wr = wid & 3, wc = wid >> 2;
    int hi = lane >> 4, ln = lane & 15;

    __shared__ __align__(16) bf16_t As[2][128 * (ROWB / 2)];
    __shared__ __align__(16) bf16_t Bs[2][64 * (ROWB / 2)];

    int ar = tid >> 1, ah = tid & 1;
    int arow_g = m0 + ar;
    if (arow_g >= cnt) arow_g = cnt - 1;
    const bf16_t* aSrc = h + (size_t)(off + arow_g) * KF + ah * 32;
    unsigned aDst0 = lds_off(&As[0][0]) + ar * ROWB + ah * 64;
    unsigned aDst1 = lds_off(&As[1][0]) + ar * ROWB + ah * 64;
    int br = tid >> 2, bq = tid & 3;
    const bf16_t* bSrc = w2t + ((size_t)e * KD + (n0 + br)) * KF + bq * 16;
    unsigned bDst0 = lds_off(&Bs[0][0]) + br * ROWB + bq * 32;
    unsigned bDst1 = lds_off(&Bs[1][0]) + br * ROWB + bq * 32;

    auto issue_stage = [&](int k0, int b) {
        unsigned ad = b ? aDst1 : aDst0;
        unsigned bd = b ? bDst1 : bDst0;
#pragma unroll
        for (int q = 0; q < 4; q++) async_ld16(ad + q * 16, aSrc + k0 + q * 8);
#pragma unroll
        for (int q = 0; q < 2; q++) async_ld16(bd + q * 16, bSrc + k0 + q * 8);
    };

    v8f c[2][2];
#pragma unroll
    for (int i = 0; i < 2; i++)
#pragma unroll
        for (int j = 0; j < 2; j++) c[i][j] = (v8f)(0.0f);

    const int NST = KF / 64;
    issue_stage(0, 0);
    for (int s = 0; s < NST; s++) {
        __syncthreads();
        if (s + 1 < NST) {
            issue_stage((s + 1) * 64, (s + 1) & 1);
            wait_asynccnt<6>();
        } else {
            wait_asynccnt<0>();
        }
        __syncthreads();
        const bf16_t* Ab = As[s & 1];
        const bf16_t* Bb = Bs[s & 1];
#pragma unroll
        for (int kc = 0; kc < 2; kc++) {
            int kc2 = kc * 64;
            v16bf a[2], b[2];
#pragma unroll
            for (int i = 0; i < 2; i++) a[i] = lds_frag_a(Ab, wr * 32 + i * 16 + ln, kc2, hi);
#pragma unroll
            for (int j = 0; j < 2; j++) b[j] = lds_frag_b(Bb, wc * 32 + j * 16 + ln, kc2, hi);
#pragma unroll
            for (int i = 0; i < 2; i++)
#pragma unroll
                for (int j = 0; j < 2; j++)
                    c[i][j] = __builtin_amdgcn_wmma_f32_16x16x32_bf16(
                        false, a[i], false, b[j], (short)0, c[i][j], false, false);
        }
    }

#pragma unroll
    for (int j = 0; j < 2; j++) {
        int nd = n0 + wc * 32 + j * 16 + ln;
#pragma unroll
        for (int i = 0; i < 2; i++) {
#pragma unroll
            for (int k = 0; k < 8; k++) {
                int mg = m0 + wr * 32 + i * 16 + k + hi * 8;
                if (mg < cnt) y[(size_t)(off + mg) * KD + nd] = c[i][j][k];
            }
        }
    }
}

// -------- combine: out[t] = w0*(y[s0]+b2[e0]) + w1*(y[s1]+b2[e1]) --------
__global__ __launch_bounds__(256) void combine_kernel(const float* __restrict__ y,
                                                      const float* __restrict__ b2,
                                                      const int* __restrict__ topk_idx,
                                                      const float* __restrict__ topk_w,
                                                      const int* __restrict__ slot_of,
                                                      float* __restrict__ out) {
    int t = blockIdx.x;
    int s0 = slot_of[2 * t], s1 = slot_of[2 * t + 1];
    int e0 = topk_idx[2 * t], e1 = topk_idx[2 * t + 1];
    float w0 = topk_w[2 * t], w1 = topk_w[2 * t + 1];
    const float4* y0 = (const float4*)(y + (size_t)s0 * KD);
    const float4* y1 = (const float4*)(y + (size_t)s1 * KD);
    const float4* bb0 = (const float4*)(b2 + (size_t)e0 * KD);
    const float4* bb1 = (const float4*)(b2 + (size_t)e1 * KD);
    float4* o = (float4*)(out + (size_t)t * KD);
    for (int c = threadIdx.x; c < KD / 4; c += 256) {
        float4 a = y0[c], b = bb0[c], cc = y1[c], d = bb1[c];
        float4 r;
        r.x = w0 * (a.x + b.x) + w1 * (cc.x + d.x);
        r.y = w0 * (a.y + b.y) + w1 * (cc.y + d.y);
        r.z = w0 * (a.z + b.z) + w1 * (cc.z + d.z);
        r.w = w0 * (a.w + b.w) + w1 * (cc.w + d.w);
        o[c] = r;
    }
}

extern "C" void kernel_launch(void* const* d_in, const int* in_sizes, int n_in,
                              void* d_out, int out_size, void* d_ws, size_t ws_size,
                              hipStream_t stream) {
    const float* x        = (const float*)d_in[0];
    const float* router_w = (const float*)d_in[1];
    const float* router_b = (const float*)d_in[2];
    const float* w1       = (const float*)d_in[3];
    const float* b1       = (const float*)d_in[4];
    const float* w3       = (const float*)d_in[5];
    const float* b3       = (const float*)d_in[6];
    const float* w2       = (const float*)d_in[7];
    const float* b2       = (const float*)d_in[8];
    float* out = (float*)d_out;

    char* ws = (char*)d_ws;
    size_t o = 0;
    auto carve = [&](size_t bytes) -> char* {
        char* p = ws + o;
        o = (o + bytes + 255) & ~(size_t)255;
        return p;
    };
    bf16_t* xb      = (bf16_t*)carve((size_t)KT * KD * 2);
    bf16_t* w1t     = (bf16_t*)carve((size_t)KE * KF * KD * 2);
    bf16_t* w3t     = (bf16_t*)carve((size_t)KE * KF * KD * 2);
    bf16_t* w2t     = (bf16_t*)carve((size_t)KE * KD * KF * 2);
    bf16_t* hbuf    = (bf16_t*)carve((size_t)KT * KK * KF * 2);
    float*  ybuf    = (float*)carve((size_t)KT * KK * KD * 4);
    int*    topkI   = (int*)carve((size_t)KT * KK * 4);
    float*  topkW   = (float*)carve((size_t)KT * KK * 4);
    int*    perm    = (int*)carve((size_t)KT * KK * 4);
    int*    slot_of = (int*)carve((size_t)KT * KK * 4);
    int*    counts  = (int*)carve(KE * 4);
    int*    offsets = (int*)carve(KE * 4);

    // 1. precision conversion (+ transposes so WMMA B fragments load contiguously)
    cvtx_kernel<<<(KT * KD) / (256 * 8), 256, 0, stream>>>(x, xb);
    transcvt_kernel<<<dim3(KF / 32, KD / 32, KE), 256, 0, stream>>>(w1, w1t, KD, KF);
    transcvt_kernel<<<dim3(KF / 32, KD / 32, KE), 256, 0, stream>>>(w3, w3t, KD, KF);
    transcvt_kernel<<<dim3(KD / 32, KF / 32, KE), 256, 0, stream>>>(w2, w2t, KF, KD);

    // 2. routing + deterministic stable sort of slots by expert
    router_kernel<<<KT / 8, 256, 0, stream>>>(x, router_w, router_b, topkI, topkW);
    sortk_kernel<<<1, 256, 0, stream>>>(topkI, perm, slot_of, counts, offsets);

    // 3. expert GEMMs (sparse: only top-2 slots computed), bf16 WMMA, f32 accum,
    //    async double-buffered LDS staging
    gemm1_kernel<<<dim3(KF / 64, (KT * KK) / 128, KE), 256, 0, stream>>>(
        xb, w1t, w3t, b1, b3, counts, offsets, perm, hbuf);
    gemm2_kernel<<<dim3(KD / 64, (KT * KK) / 128, KE), 256, 0, stream>>>(
        hbuf, w2t, counts, offsets, ybuf);

    // 4. weighted combine with b2
    combine_kernel<<<KT, 256, 0, stream>>>(ybuf, b2, topkI, topkW, slot_of, out);
}
